// PointNet2SemSeg_12103217840689
// MI455X (gfx1250) — compile-verified, matrix operand order checked by reading, not run on hardware
//
#include <hip/hip_runtime.h>
#include <hip/hip_bf16.h>
#include <math.h>

// ---------------------------------------------------------------------------
// PointNet++ MSG semantic segmentation forward for MI455X (gfx1250).
// All linear layers run through v_wmma_f32_16x16x32_f16 with bias+BN+ReLU
// fused in the epilogue.  K is padded to a multiple of 32 everywhere
// (padded + pre-transposed f16 weights, padded activation strides) so the
// GEMM is a single unpredicated fast path: 4x b128 A loads + 1x b128 B load
// per thread per K-step, 6x ds_load_b128 fragment loads, 2 WMMAs per wave.
// ---------------------------------------------------------------------------

typedef __attribute__((ext_vector_type(16))) _Float16 v16h;
typedef __attribute__((ext_vector_type(8)))  _Float16 v8h;
typedef __attribute__((ext_vector_type(8)))  float    v8f;

#define BN_INV_F 0.9999950000374997f   // 1/sqrt(1+1e-5)

static inline int pad32(int x) { return (x + 31) & ~31; }

#define GM 64
#define GN 32
#define GK 32
#define SA_LD 40   // halves per row (80B: 16B-aligned rows -> ds_load_b128)
#define SB_LD 40

// Preconditions: M % 64 == 0, K % 32 == 0 (by construction).
// WT is the weight pre-transposed: N rows x K cols (K = padded), f16.
__global__ __launch_bounds__(128) void wmma_linear_kernel(
    const float* __restrict__ A, int lda, long strideA,
    const _Float16* __restrict__ WT,
    const float* __restrict__ bias,
    const float* __restrict__ gamma,
    const float* __restrict__ beta,
    float* __restrict__ D, int ldd, long strideD,
    int M, int N, int K, int relu, int Nz)
{
    __shared__ _Float16 sA[GM * SA_LD];   // 64x32 A tile (f16)
    __shared__ _Float16 sB[GN * SB_LD];   // 32(n) x 32(k) B tile, (n,k) layout

    A += (long)blockIdx.z * strideA;
    D += (long)blockIdx.z * strideD;

    const int t    = threadIdx.x;
    const int wave = t >> 5;
    const int lane = t & 31;
    const int row0 = blockIdx.x * GM;
    const int col0 = blockIdx.y * GN;

    v8f acc0 = {0.f, 0.f, 0.f, 0.f, 0.f, 0.f, 0.f, 0.f};
    v8f acc1 = {0.f, 0.f, 0.f, 0.f, 0.f, 0.f, 0.f, 0.f};

    // A-load mapping: thread -> (row, 16-wide K chunk)
    const int ar  = t >> 1;
    const int aks = (t & 1) << 4;             // 0 or 16
    // B-load mapping: thread -> (col, 8-wide K chunk)
    const int bn  = t >> 2;
    const int bks = (t & 3) << 3;             // 0,8,16,24
    const int bnn = col0 + bn;

    for (int k0 = 0; k0 < K; k0 += GK) {
        // ---- A tile: unpredicated batched loads, f32 -> f16
        {
            const float* src = A + (long)(row0 + ar) * lda + k0 + aks;
            v8h lo, hi;
#pragma unroll
            for (int j = 0; j < 8; ++j) {
                lo[j] = (_Float16)src[j];
                hi[j] = (_Float16)src[j + 8];
            }
            *(v8h*)&sA[ar * SA_LD + aks]     = lo;
            *(v8h*)&sA[ar * SA_LD + aks + 8] = hi;
        }
        // ---- B tile: WT[n][k] (already transposed) -> one aligned b128 load
        {
            v8h v = {};
            if (bnn < N) v = *(const v8h*)(WT + (long)bnn * K + k0 + bks);
            *(v8h*)&sB[bn * SB_LD + bks] = v;
        }
        __syncthreads();

        // Fragments per ISA 16-bit 16x32 layout: lanes 0-15 hold row M=lane,
        // halves 0..7 = K0..7, 8..15 = K16..23; lanes 16-31 shift K by +8.
        const _Float16* pa  = &sA[(wave * 16 + (lane & 15)) * SA_LD + ((lane >> 4) << 3)];
        const _Float16* pb0 = &sB[(lane & 15) * SB_LD + ((lane >> 4) << 3)];
        const _Float16* pb1 = pb0 + 16 * SB_LD;
        v16h af, bf0, bf1;
#pragma unroll
        for (int h = 0; h < 8; ++h) {
            af[h]  = pa[h];   af[h + 8]  = pa[h + 16];
            bf0[h] = pb0[h];  bf0[h + 8] = pb0[h + 16];
            bf1[h] = pb1[h];  bf1[h + 8] = pb1[h + 16];
        }
        acc0 = __builtin_amdgcn_wmma_f32_16x16x32_f16(
                   false, af, false, bf0, (short)0, acc0, false, false);
        acc1 = __builtin_amdgcn_wmma_f32_16x16x32_f16(
                   false, af, false, bf1, (short)0, acc1, false, false);
        __syncthreads();
    }

    // Epilogue: VGPR r holds M = r (lanes 0-15) / 8+r (lanes 16-31), N = lane&15
    int colA  = col0 + (lane & 15);
    int colB  = colA + 16;
    int mbase = row0 + wave * 16 + ((lane >> 4) << 3);
    float bbA = 0.f, scA = 1.f, shA = 0.f, bbB = 0.f, scB = 1.f, shB = 0.f;
    if (colA < N) {
        bbA = bias ? bias[colA] : 0.f;
        scA = gamma ? gamma[colA] * BN_INV_F : 1.f;
        shA = beta ? beta[colA] : 0.f;
    }
    if (colB < N) {
        bbB = bias ? bias[colB] : 0.f;
        scB = gamma ? gamma[colB] * BN_INV_F : 1.f;
        shB = beta ? beta[colB] : 0.f;
    }
#pragma unroll
    for (int r = 0; r < 8; ++r) {
        int m = mbase + r;
        if (colA < N) {
            float v = (acc0[r] + bbA) * scA + shA;
            if (relu) v = fmaxf(v, 0.0f);
            D[(long)m * ldd + colA] = v;
        } else if (colA < Nz) {
            D[(long)m * ldd + colA] = 0.0f;     // zero-fill K padding for next layer
        }
        if (colB < N) {
            float v = (acc1[r] + bbB) * scB + shB;
            if (relu) v = fmaxf(v, 0.0f);
            D[(long)m * ldd + colB] = v;
        } else if (colB < Nz) {
            D[(long)m * ldd + colB] = 0.0f;
        }
    }
}

// ------------------------------ helpers ------------------------------------

// Convert W (K x N, f32) to transposed f16 WT (N x Kpad), zero pad k >= K.
__global__ void cvt_pad_wT_kernel(const float* __restrict__ src, _Float16* __restrict__ dst,
                                  int K, int N, int Kpad)
{
    long i = blockIdx.x * (long)blockDim.x + threadIdx.x;
    long tot = (long)N * Kpad;
    if (i >= tot) return;
    int k = (int)(i % Kpad);
    int n = (int)(i / Kpad);
    dst[i] = (k < K) ? (_Float16)src[(long)k * N + n] : (_Float16)0.0f;
}

// (B,C,N) -> (B,N,Cp) with zero pad for c >= C.
__global__ void transpose_pad_kernel(const float* __restrict__ src, float* __restrict__ dst,
                                     int B, int C, int N, int Cp)
{
    long i = blockIdx.x * (long)blockDim.x + threadIdx.x;
    long tot = (long)B * N * Cp;
    if (i >= tot) return;
    int c = (int)(i % Cp);
    long r = i / Cp;
    int n = (int)(r % N);
    int b = (int)(r / N);
    dst[i] = (c < C) ? src[((long)b * C + c) * N + n] : 0.0f;
}

// Farthest point sampling: one block per batch; dist kept in LDS (N<=4096).
__global__ void fps_kernel(const float* __restrict__ xyz, int N, int npoint,
                           float* __restrict__ new_xyz)
{
    __shared__ float dist[4096];
    __shared__ float rv[256];
    __shared__ int   ri[256];
    int b = blockIdx.x;
    int t = threadIdx.x;
    const float* X = xyz + (long)b * N * 3;
    float* NX = new_xyz + (long)b * npoint * 3;

    for (int i = t; i < N; i += 256) dist[i] = 1e10f;
    __syncthreads();

    int last = 0;
    for (int s = 0; s < npoint; ++s) {
        float lx = X[last * 3 + 0], ly = X[last * 3 + 1], lz = X[last * 3 + 2];
        if (t == 0) { NX[s * 3 + 0] = lx; NX[s * 3 + 1] = ly; NX[s * 3 + 2] = lz; }
        float bv = -1.0f; int bi = 0;
        for (int i = t; i < N; i += 256) {
            float dx = X[i * 3 + 0] - lx, dy = X[i * 3 + 1] - ly, dz = X[i * 3 + 2] - lz;
            float d  = dx * dx + dy * dy + dz * dz;
            float nd = fminf(dist[i], d);
            dist[i] = nd;
            if (nd > bv) { bv = nd; bi = i; }
        }
        rv[t] = bv; ri[t] = bi;
        __syncthreads();
        for (int off = 128; off > 0; off >>= 1) {
            if (t < off) {
                if (rv[t + off] > rv[t] ||
                    (rv[t + off] == rv[t] && ri[t + off] < ri[t])) {
                    rv[t] = rv[t + off]; ri[t] = ri[t + off];
                }
            }
            __syncthreads();
        }
        last = ri[0];
        __syncthreads();
    }
}

// Ball query: first `ns` ascending indices with d2 < r2 (== sort semantics),
// pad with first hit. grid: (ceil(npoint/256), B).
__global__ void ball_query_kernel(float r2, int ns, const float* __restrict__ xyz,
                                  const float* __restrict__ nxyz, int N, int npoint,
                                  int* __restrict__ idx)
{
    int q = blockIdx.x * blockDim.x + threadIdx.x;
    int b = blockIdx.y;
    if (q >= npoint) return;
    const float* X = xyz + (long)b * N * 3;
    const float* Q = nxyz + ((long)b * npoint + q) * 3;
    int* out = idx + ((long)b * npoint + q) * ns;
    float qx = Q[0], qy = Q[1], qz = Q[2];
    int cnt = 0;
    for (int n = 0; n < N && cnt < ns; ++n) {
        float dx = X[n * 3 + 0] - qx, dy = X[n * 3 + 1] - qy, dz = X[n * 3 + 2] - qz;
        if (dx * dx + dy * dy + dz * dz < r2) out[cnt++] = n;
    }
    int f = (cnt > 0) ? out[0] : 0;
    for (int k = cnt; k < ns; ++k) out[k] = f;
}

// Grouping: rows (npoint*ns) x Kpad = [rel_xyz | feats | 0 pad].
// Per-batch launch (pointers pre-offset).
__global__ void group_kernel(const float* __restrict__ xyz, const float* __restrict__ feats,
                             int featC, int fstride, const float* __restrict__ nxyz,
                             const int* __restrict__ idx, int N, int npoint, int ns,
                             int Kpad, float* __restrict__ Aout)
{
    long i = blockIdx.x * (long)blockDim.x + threadIdx.x;
    long tot = (long)npoint * ns * Kpad;
    if (i >= tot) return;
    int c = (int)(i % Kpad);
    long r = i / Kpad;
    int s = (int)(r % ns);
    int q = (int)(r / ns);
    int pi = idx[(long)q * ns + s];
    if (pi >= N) pi = N - 1;  // jnp OOB clamp
    float v;
    if (c < 3)                v = xyz[pi * 3 + c] - nxyz[q * 3 + c];
    else if (c < 3 + featC)   v = feats[(long)pi * fstride + (c - 3)];
    else                      v = 0.0f;
    Aout[i] = v;
}

// Max over the ns sample axis (H row stride ldh); writes at channel offset.
__global__ void maxpool_kernel(const float* __restrict__ H, int npoint, int ns, int C,
                               int ldh, float* __restrict__ dst, int dstride, int choff)
{
    int i = blockIdx.x * blockDim.x + threadIdx.x;
    if (i >= npoint * C) return;
    int c = i % C;
    int q = i / C;
    const float* p = H + ((long)q * ns) * ldh + c;
    float m = p[0];
    for (int s = 1; s < ns; ++s) m = fmaxf(m, p[(long)s * ldh]);
    dst[(long)q * dstride + choff + c] = m;
}

// 3-NN (stable ties == lax.top_k) + normalized inverse-distance weights.
__global__ void knn3_kernel(const float* __restrict__ uxyz, const float* __restrict__ kxyz,
                            int nU, int nK, int* __restrict__ idx3, float* __restrict__ w3)
{
    int u = blockIdx.x * blockDim.x + threadIdx.x;
    int b = blockIdx.y;
    if (u >= nU) return;
    const float* U  = uxyz + ((long)b * nU + u) * 3;
    const float* Kx = kxyz + (long)b * nK * 3;
    float ux = U[0], uy = U[1], uz = U[2];
    float d0 = 1e30f, d1 = 1e30f, d2 = 1e30f;
    int i0 = 0, i1 = 0, i2 = 0;
    for (int k = 0; k < nK; ++k) {
        float dx = Kx[k * 3 + 0] - ux, dy = Kx[k * 3 + 1] - uy, dz = Kx[k * 3 + 2] - uz;
        float d = dx * dx + dy * dy + dz * dz;
        if (d < d0)      { d2 = d1; i2 = i1; d1 = d0; i1 = i0; d0 = d; i0 = k; }
        else if (d < d1) { d2 = d1; i2 = i1; d1 = d;  i1 = k; }
        else if (d < d2) { d2 = d;  i2 = k; }
    }
    float w0 = 1.0f / (d0 + 1e-8f), w1 = 1.0f / (d1 + 1e-8f), w2 = 1.0f / (d2 + 1e-8f);
    float s = w0 + w1 + w2;
    long o = ((long)b * nU + u) * 3;
    idx3[o] = i0; idx3[o + 1] = i1; idx3[o + 2] = i2;
    w3[o] = w0 / s; w3[o + 1] = w1 / s; w3[o + 2] = w2 / s;
}

// cat = [interp(kfeat) | ufeat], shape (B, nU, Ck+Cu). grid.y = B.
__global__ void interp_concat_kernel(const float* __restrict__ kfeat, int Ck, long kb,
                                     const float* __restrict__ ufeat, int Cu, long ub,
                                     const int* __restrict__ idx3, const float* __restrict__ w3,
                                     float* __restrict__ cat, int nU)
{
    int Ct = Ck + Cu;
    long i = blockIdx.x * (long)blockDim.x + threadIdx.x;
    int b = blockIdx.y;
    if (i >= (long)nU * Ct) return;
    int c = (int)(i % Ct);
    int u = (int)(i / Ct);
    float v;
    if (c < Ck) {
        long o = ((long)b * nU + u) * 3;
        const float* KF = kfeat + (long)b * kb;
        v = w3[o]     * KF[(long)idx3[o]     * Ck + c]
          + w3[o + 1] * KF[(long)idx3[o + 1] * Ck + c]
          + w3[o + 2] * KF[(long)idx3[o + 2] * Ck + c];
    } else {
        v = ufeat[(long)b * ub + (long)u * Cu + (c - Ck)];
    }
    cat[(long)b * nU * Ct + i] = v;
}

// l3 *= sigmoid(score) broadcast over channels.
__global__ void gate_kernel(float* __restrict__ x, const float* __restrict__ s,
                            long tot, int C)
{
    long i = blockIdx.x * (long)blockDim.x + threadIdx.x;
    if (i >= tot) return;
    long q = i / C;
    x[i] *= 1.0f / (1.0f + expf(-s[q]));
}

// --------------------------- host orchestration ----------------------------

struct Lyr { const float *W, *b, *be, *g; int cin, cout, Kp; _Float16* Wh; };

static void gemm_launch(hipStream_t s, const float* A, int lda, long strideA,
                        const _Float16* WT, const float* bias, const float* g,
                        const float* be, float* D, int ldd, long strideD,
                        int M, int Ncols, int K, int relu, int Nz, int batches)
{
    dim3 grid((unsigned)(M / GM), (unsigned)((Ncols + GN - 1) / GN), (unsigned)batches);
    wmma_linear_kernel<<<grid, 128, 0, s>>>(A, lda, strideA, WT, bias, g, be,
                                            D, ldd, strideD, M, Ncols, K, relu, Nz);
}

static void run_sa(hipStream_t stream, int B,
                   const float* in_xyz, int Nin, const float* in_feats, int inC,
                   int fstride, int npoint, float* out_xyz, float* out_feats,
                   int outCtot, int nbranch, const float* radii, const int* nsamp,
                   Lyr* mlps, int* ballidx, float* Agroup, float* H0, float* H1)
{
    fps_kernel<<<B, 256, 0, stream>>>(in_xyz, Nin, npoint, out_xyz);
    int choff = 0;
    for (int j = 0; j < nbranch; ++j) {
        float r = radii[j];
        int ns = nsamp[j];
        dim3 bg((unsigned)((npoint + 255) / 256), (unsigned)B);
        ball_query_kernel<<<bg, 256, 0, stream>>>(r * r, ns, in_xyz, out_xyz,
                                                  Nin, npoint, ballidx);
        int Kp0 = mlps[j * 3 + 0].Kp;   // pad32(3+inC)
        int Mr = npoint * ns;
        for (int b = 0; b < B; ++b) {
            long tot = (long)Mr * Kp0;
            group_kernel<<<(unsigned)((tot + 255) / 256), 256, 0, stream>>>(
                in_xyz + (long)b * Nin * 3, in_feats + (long)b * Nin * fstride, inC,
                fstride, out_xyz + (long)b * npoint * 3,
                ballidx + (long)b * npoint * ns, Nin, npoint, ns, Kp0, Agroup);
            const float* src = Agroup;
            float* dsts[3] = {H0, H1, H0};
            for (int l = 0; l < 3; ++l) {
                Lyr& L = mlps[j * 3 + l];
                int op = pad32(L.cout);
                gemm_launch(stream, src, L.Kp, 0, L.Wh, L.b, L.g, L.be,
                            dsts[l], op, 0, Mr, L.cout, L.Kp, 1, op, 1);
                src = dsts[l];
            }
            int cout = mlps[j * 3 + 2].cout;
            int tot2 = npoint * cout;
            maxpool_kernel<<<(unsigned)((tot2 + 255) / 256), 256, 0, stream>>>(
                src, npoint, ns, cout, pad32(cout),
                out_feats + (long)b * npoint * outCtot, outCtot, choff);
        }
        choff += mlps[j * 3 + 2].cout;
    }
}

extern "C" void kernel_launch(void* const* d_in, const int* in_sizes, int n_in,
                              void* d_out, int out_size, void* d_ws, size_t ws_size,
                              hipStream_t stream)
{
    (void)in_sizes; (void)n_in; (void)out_size; (void)ws_size;
    const int B = 8, Np = 4096, Cf = 8;

    const float* xyz      = (const float*)d_in[0];
    const float* features = (const float*)d_in[1];

    // params flattened pytree-style (dict keys sorted):
    // attention, classifier, conv, fp1, fp2, fp3, sa1, sa2, sa3
    int p = 2;
    auto F = [&](int i) { return (const float*)d_in[i]; };
    const float *att_W1 = F(p++), *att_W2 = F(p++), *att_b1 = F(p++),
                *att_b2 = F(p++), *att_be1 = F(p++), *att_g1 = F(p++);
    const float *cls_W1 = F(p++), *cls_W2 = F(p++), *cls_b1 = F(p++),
                *cls_b2 = F(p++), *cls_be = F(p++), *cls_g = F(p++);
    const float *conv_W = F(p++), *conv_b = F(p++);

    auto readL = [&](int cin, int cout) -> Lyr {
        Lyr L; L.W = F(p++); L.b = F(p++); L.be = F(p++); L.g = F(p++);
        L.cin = cin; L.cout = cout; L.Kp = pad32(cin); L.Wh = nullptr; return L;
    };
    Lyr fp1L[2] = { readL(320, 128),  readL(128, 128) };
    Lyr fp2L[2] = { readL(832, 256),  readL(256, 256) };
    Lyr fp3L[2] = { readL(1664, 512), readL(512, 512) };
    int sa1d[3][4] = {{11, 32, 32, 64}, {11, 64, 64, 128}, {11, 64, 96, 128}};
    int sa2d[3][4] = {{323, 64, 64, 128}, {323, 128, 128, 256}, {323, 128, 196, 256}};
    int sa3d[4]    = {643, 256, 512, 1024};
    Lyr sa1L[9], sa2L[9], sa3L[3];
    for (int m = 0; m < 3; ++m)
        for (int l = 0; l < 3; ++l) sa1L[m * 3 + l] = readL(sa1d[m][l], sa1d[m][l + 1]);
    for (int m = 0; m < 3; ++m)
        for (int l = 0; l < 3; ++l) sa2L[m * 3 + l] = readL(sa2d[m][l], sa2d[m][l + 1]);
    for (int l = 0; l < 3; ++l) sa3L[l] = readL(sa3d[l], sa3d[l + 1]);

    // ----- workspace carve -----
    char* wsb = (char*)d_ws;
    size_t off = 0;
    auto alloc = [&](size_t bytes) -> void* {
        void* r = wsb + off; off += (bytes + 255) & ~(size_t)255; return r;
    };
    float* featsP   = (float*)alloc((size_t)B * Np * 32 * 4);       // padded to 32 ch
    float* residual = (float*)alloc((size_t)B * Np * 64 * 4);
    float* l1_xyz   = (float*)alloc((size_t)B * 1024 * 3 * 4);
    float* l1       = (float*)alloc((size_t)B * 1024 * 320 * 4);
    float* l2_xyz   = (float*)alloc((size_t)B * 256 * 3 * 4);
    float* l2       = (float*)alloc((size_t)B * 256 * 640 * 4);
    float* l3_xyz   = (float*)alloc((size_t)B * 64 * 3 * 4);
    float* l3       = (float*)alloc((size_t)B * 64 * 1024 * 4);
    float* att_h    = (float*)alloc((size_t)B * 64 * 512 * 4);
    float* att_s    = (float*)alloc((size_t)B * 64 * 4);
    float* fp3cat   = (float*)alloc((size_t)B * 256 * 1664 * 4);
    float* fp3h     = (float*)alloc((size_t)B * 256 * 512 * 4);
    float* fp3o     = (float*)alloc((size_t)B * 256 * 512 * 4);
    float* fp2cat   = (float*)alloc((size_t)B * 1024 * 832 * 4);
    float* fp2h     = (float*)alloc((size_t)B * 1024 * 256 * 4);
    float* fp2o     = (float*)alloc((size_t)B * 1024 * 256 * 4);
    float* fp1cat   = (float*)alloc((size_t)B * 4096 * 320 * 4);
    float* fp1h     = (float*)alloc((size_t)B * 4096 * 128 * 4);
    float* fp1o     = (float*)alloc((size_t)B * 4096 * 128 * 4);
    float* clsh     = (float*)alloc((size_t)B * 4096 * 128 * 4);
    int*   ballidx  = (int*)alloc((size_t)B * 1024 * 64 * 4);
    int*   knn_i    = (int*)alloc((size_t)B * 4096 * 3 * 4);
    float* knn_w    = (float*)alloc((size_t)B * 4096 * 3 * 4);
    float* Agroup   = (float*)alloc((size_t)256 * 128 * 352 * 4);   // per-batch max (Kp)
    float* H0       = (float*)alloc((size_t)8388608 * 4);
    float* H1       = (float*)alloc((size_t)8388608 * 4);
    _Float16* wpool = (_Float16*)alloc((size_t)4 * 1024 * 1024 * 2);

    // ----- weight f32 -> transposed padded f16 (deterministic, every call) --
    size_t woff = 0;
    auto cvtW = [&](const float* src, int K, int N) -> _Float16* {
        int Kp = pad32(K);
        long n = (long)N * Kp;
        _Float16* d = wpool + woff;
        woff += (size_t)((n + 127) & ~127);
        cvt_pad_wT_kernel<<<(unsigned)((n + 255) / 256), 256, 0, stream>>>(src, d, K, N, Kp);
        return d;
    };
    _Float16* convWh = cvtW(conv_W, 8, 64);
    _Float16* attW1h = cvtW(att_W1, 1024, 512);
    _Float16* attW2h = cvtW(att_W2, 512, 1);
    _Float16* clsW1h = cvtW(cls_W1, 128, 128);
    _Float16* clsW2h = cvtW(cls_W2, 128, 2);
    for (int i = 0; i < 2; ++i) { fp1L[i].Wh = cvtW(fp1L[i].W, fp1L[i].cin, fp1L[i].cout); }
    for (int i = 0; i < 2; ++i) { fp2L[i].Wh = cvtW(fp2L[i].W, fp2L[i].cin, fp2L[i].cout); }
    for (int i = 0; i < 2; ++i) { fp3L[i].Wh = cvtW(fp3L[i].W, fp3L[i].cin, fp3L[i].cout); }
    for (int i = 0; i < 9; ++i) { sa1L[i].Wh = cvtW(sa1L[i].W, sa1L[i].cin, sa1L[i].cout); }
    for (int i = 0; i < 9; ++i) { sa2L[i].Wh = cvtW(sa2L[i].W, sa2L[i].cin, sa2L[i].cout); }
    for (int i = 0; i < 3; ++i) { sa3L[i].Wh = cvtW(sa3L[i].W, sa3L[i].cin, sa3L[i].cout); }

    // ----- pipeline -----
    {   // featsP = transpose(features) padded to 32; residual = featsP @ convW + b
        long tot = (long)B * Np * 32;
        transpose_pad_kernel<<<(unsigned)((tot + 255) / 256), 256, 0, stream>>>(
            features, featsP, B, Cf, Np, 32);
        gemm_launch(stream, featsP, 32, (long)Np * 32, convWh, conv_b, nullptr, nullptr,
                    residual, 64, (long)Np * 64, Np, 64, 32, 0, 64, B);
    }

    {   // SA1: 4096 -> 1024, 3 branches
        float radii[3] = {0.05f, 0.1f, 0.2f};
        int   nsv[3]   = {16, 32, 64};
        run_sa(stream, B, xyz, Np, featsP, Cf, 32, 1024, l1_xyz, l1, 320,
               3, radii, nsv, sa1L, ballidx, Agroup, H0, H1);
    }
    {   // SA2: 1024 -> 256
        float radii[3] = {0.2f, 0.4f, 0.8f};
        int   nsv[3]   = {32, 64, 128};
        run_sa(stream, B, l1_xyz, 1024, l1, 320, 320, 256, l2_xyz, l2, 640,
               3, radii, nsv, sa2L, ballidx, Agroup, H0, H1);
    }
    {   // SA3: 256 -> 64, single branch
        float radii[1] = {1.2f};
        int   nsv[1]   = {128};
        run_sa(stream, B, l2_xyz, 256, l2, 640, 640, 64, l3_xyz, l3, 1024,
               1, radii, nsv, sa3L, ballidx, Agroup, H0, H1);
    }

    // attention gating on l3
    gemm_launch(stream, l3, 1024, (long)64 * 1024, attW1h, att_b1, att_g1, att_be1,
                att_h, 512, (long)64 * 512, 64, 512, 1024, 1, 512, B);
    gemm_launch(stream, att_h, 512, (long)64 * 512, attW2h, att_b2, nullptr, nullptr,
                att_s, 1, (long)64, 64, 1, 512, 0, 1, B);
    {
        long tot = (long)B * 64 * 1024;
        gate_kernel<<<(unsigned)((tot + 255) / 256), 256, 0, stream>>>(l3, att_s, tot, 1024);
    }

    // FP3: (l2_xyz <- l3_xyz), cat 1024+640=1664 -> 512 -> 512
    knn3_kernel<<<dim3(1, B), 256, 0, stream>>>(l2_xyz, l3_xyz, 256, 64, knn_i, knn_w);
    {
        long tot = (long)256 * 1664;
        interp_concat_kernel<<<dim3((unsigned)((tot + 255) / 256), B), 256, 0, stream>>>(
            l3, 1024, (long)64 * 1024, l2, 640, (long)256 * 640, knn_i, knn_w, fp3cat, 256);
    }
    gemm_launch(stream, fp3cat, 1664, (long)256 * 1664, fp3L[0].Wh, fp3L[0].b, fp3L[0].g,
                fp3L[0].be, fp3h, 512, (long)256 * 512, 256, 512, 1664, 1, 512, B);
    gemm_launch(stream, fp3h, 512, (long)256 * 512, fp3L[1].Wh, fp3L[1].b, fp3L[1].g,
                fp3L[1].be, fp3o, 512, (long)256 * 512, 256, 512, 512, 1, 512, B);

    // FP2: (l1_xyz <- l2_xyz), cat 512+320=832 -> 256 -> 256
    knn3_kernel<<<dim3(4, B), 256, 0, stream>>>(l1_xyz, l2_xyz, 1024, 256, knn_i, knn_w);
    {
        long tot = (long)1024 * 832;
        interp_concat_kernel<<<dim3((unsigned)((tot + 255) / 256), B), 256, 0, stream>>>(
            fp3o, 512, (long)256 * 512, l1, 320, (long)1024 * 320, knn_i, knn_w, fp2cat, 1024);
    }
    gemm_launch(stream, fp2cat, 832, (long)1024 * 832, fp2L[0].Wh, fp2L[0].b, fp2L[0].g,
                fp2L[0].be, fp2h, 256, (long)1024 * 256, 1024, 256, 832, 1, 256, B);
    gemm_launch(stream, fp2h, 256, (long)1024 * 256, fp2L[1].Wh, fp2L[1].b, fp2L[1].g,
                fp2L[1].be, fp2o, 256, (long)1024 * 256, 1024, 256, 256, 1, 256, B);

    // FP1: (xyz <- l1_xyz), cat 256+64=320 -> 128 -> 128
    knn3_kernel<<<dim3(16, B), 256, 0, stream>>>(xyz, l1_xyz, 4096, 1024, knn_i, knn_w);
    {
        long tot = (long)4096 * 320;
        interp_concat_kernel<<<dim3((unsigned)((tot + 255) / 256), B), 256, 0, stream>>>(
            fp2o, 256, (long)1024 * 256, residual, 64, (long)4096 * 64, knn_i, knn_w,
            fp1cat, 4096);
    }
    gemm_launch(stream, fp1cat, 320, (long)4096 * 320, fp1L[0].Wh, fp1L[0].b, fp1L[0].g,
                fp1L[0].be, fp1h, 128, (long)4096 * 128, 4096, 128, 320, 1, 128, B);
    gemm_launch(stream, fp1h, 128, (long)4096 * 128, fp1L[1].Wh, fp1L[1].b, fp1L[1].g,
                fp1L[1].be, fp1o, 128, (long)4096 * 128, 4096, 128, 128, 1, 128, B);

    // classifier: relu BN linear then final 2-way head straight into d_out
    gemm_launch(stream, fp1o, 128, (long)4096 * 128, clsW1h, cls_b1, cls_g, cls_be,
                clsh, 128, (long)4096 * 128, 4096, 128, 128, 1, 128, B);
    gemm_launch(stream, clsh, 128, (long)4096 * 128, clsW2h, cls_b2, nullptr, nullptr,
                (float*)d_out, 2, (long)4096 * 2, 4096, 2, 128, 0, 2, B);
}
